// SetAbstraction_39316130627920
// MI455X (gfx1250) — matrix operand order repaired and checked by reading, not compiled, not run on hardware
//
#include <hip/hip_runtime.h>
#include <hip/hip_bf16.h>

// ---------- problem constants ----------
#define Bb      16
#define Nn      8192
#define NPOINT  1024
#define NSAMPLE 32
#define R2      0.04f           // RADIUS^2
#define CPAD1   96              // 67 in-channels padded to 3 K-chunks of 32
static const long long Ppos = (long long)Bb * NPOINT * NSAMPLE;   // 524288

// ---------- WMMA vector types ----------
typedef __attribute__((ext_vector_type(16))) _Float16 v16h;
typedef __attribute__((ext_vector_type(8)))  _Float16 v8h;
typedef __attribute__((ext_vector_type(8)))  float    v8f;

// =====================================================================
// 0) zero the BN statistics accumulators (atomics accumulate per call)
// =====================================================================
__global__ void zero_stats_kernel(float* stats) {
    int i = blockIdx.x * blockDim.x + threadIdx.x;
    if (i < 512) stats[i] = 0.0f;
}

// =====================================================================
// 1) Farthest point sampling: one block per batch, 1024 threads,
//    distances live in registers (8 points/thread), LDS argmax tree.
//    Writes new_xyz directly into d_out (first 16*1024*3 floats).
// =====================================================================
__global__ void __launch_bounds__(1024)
fps_kernel(const float* __restrict__ xyz, int* __restrict__ fps_idx,
           float* __restrict__ new_xyz_out) {
    __shared__ float rv[1024];
    __shared__ int   ri[1024];
    __shared__ float cc[3];
    __shared__ int   cur;

    const int b   = blockIdx.x;
    const int tid = threadIdx.x;
    const float* xb = xyz + (size_t)b * Nn * 3;

    float d[8];
#pragma unroll
    for (int i = 0; i < 8; ++i) d[i] = 1e10f;
    if (tid == 0) cur = 0;
    __syncthreads();

    for (int it = 0; it < NPOINT; ++it) {
        if (tid == 0) {
            const int c = cur;
            fps_idx[b * NPOINT + it] = c;
            const float x = xb[c * 3 + 0], y = xb[c * 3 + 1], z = xb[c * 3 + 2];
            cc[0] = x; cc[1] = y; cc[2] = z;
            float* o = new_xyz_out + ((size_t)b * NPOINT + it) * 3;
            o[0] = x; o[1] = y; o[2] = z;
        }
        __syncthreads();
        const float cx = cc[0], cy = cc[1], cz = cc[2];
        float best = -1.0f; int bj = tid;
#pragma unroll
        for (int i = 0; i < 8; ++i) {
            const int j = tid + i * 1024;
            const float dx = xb[j * 3 + 0] - cx;
            const float dy = xb[j * 3 + 1] - cy;
            const float dz = xb[j * 3 + 2] - cz;
            const float dist = dx * dx + dy * dy + dz * dz;
            d[i] = fminf(d[i], dist);
            if (d[i] > best) { best = d[i]; bj = j; }
        }
        rv[tid] = best; ri[tid] = bj;
        __syncthreads();
        for (int st = 512; st > 0; st >>= 1) {
            if (tid < st && rv[tid + st] > rv[tid]) {
                rv[tid] = rv[tid + st]; ri[tid] = ri[tid + st];
            }
            __syncthreads();
        }
        if (tid == 0) cur = ri[0];
        __syncthreads();
    }
}

// =====================================================================
// 2) Ball query + group + concat:  one wave32 per centroid.
//    Ballot/popc prefix collects the first 32 in-radius indices in
//    index order (== reference's where/sort/truncate), pads with slot0.
//    Emits X0[pos][96] f16 rows: [xyz_norm(3) | feats(64) | zeros(29)].
// =====================================================================
__global__ void __launch_bounds__(256)
ball_group_kernel(const float* __restrict__ xyz,
                  const float* __restrict__ points,
                  const float* __restrict__ new_xyz,   // in d_out
                  _Float16* __restrict__ X0) {
    __shared__ int bidx[8][NSAMPLE];
    const int wave = threadIdx.x >> 5;
    const int lane = threadIdx.x & 31;
    const int wid  = blockIdx.x * 8 + wave;        // b*NPOINT + s
    const int b    = wid >> 10;

    const float cx = new_xyz[(size_t)wid * 3 + 0];
    const float cy = new_xyz[(size_t)wid * 3 + 1];
    const float cz = new_xyz[(size_t)wid * 3 + 2];
    const float* xb = xyz + (size_t)b * Nn * 3;

    int cnt = 0;
    for (int chunk = 0; chunk < Nn / 32 && cnt < NSAMPLE; ++chunk) {
        const int j = chunk * 32 + lane;
        const float dx = xb[j * 3 + 0] - cx;
        const float dy = xb[j * 3 + 1] - cy;
        const float dz = xb[j * 3 + 2] - cz;
        const bool pred = (dx * dx + dy * dy + dz * dz) <= R2;
        const unsigned mask = (unsigned)__ballot(pred);
        if (pred) {
            const int slot = cnt + __popc(mask & ((1u << lane) - 1u));
            if (slot < NSAMPLE) bidx[wave][slot] = j;
        }
        cnt += __popc(mask);
    }
    if (cnt > NSAMPLE) cnt = NSAMPLE;
    const int first = bidx[wave][0];               // centroid itself guarantees cnt>=1
    if (lane >= cnt) bidx[wave][lane] = first;

    // gather: lane k handles sample k
    const int idx = bidx[wave][lane];
    const size_t pos = (size_t)wid * NSAMPLE + lane;
    _Float16* row = X0 + pos * CPAD1;
    row[0] = (_Float16)(xb[idx * 3 + 0] - cx);
    row[1] = (_Float16)(xb[idx * 3 + 1] - cy);
    row[2] = (_Float16)(xb[idx * 3 + 2] - cz);
    const float* fp = points + ((size_t)b * Nn + idx) * 64;
#pragma unroll 8
    for (int c = 0; c < 64; ++c) row[3 + c] = (_Float16)fp[c];
#pragma unroll
    for (int c = 67; c < CPAD1; ++c) row[c] = (_Float16)0.0f;
}

// =====================================================================
// 3) Fused (BN+ReLU on input) -> 1x1 conv GEMM via v_wmma_f32_16x16x32_f16
//    -> bias -> per-channel sum/sumsq atomics -> f16 store.
//    Block = 256 thr = 8 waves; each wave owns a 16-position N-subtile.
// =====================================================================
template <int OUTC, int CPADIN, int KC, bool APPLY_BN>
__global__ void __launch_bounds__(256)
conv_bn_wmma_kernel(const _Float16* __restrict__ Xin,
                    const float* __restrict__ W, int CIN,
                    const float* __restrict__ bias,
                    const float* __restrict__ in_scale,
                    const float* __restrict__ in_shift,
                    float* __restrict__ sum_out,
                    float* __restrict__ sq_out,
                    _Float16* __restrict__ Yout) {
    constexpr int MT = OUTC / 16;
    __shared__ __align__(32) _Float16 wlds[OUTC * CPADIN];
    __shared__ float blds[OUTC];
    __shared__ float sclds[CPADIN];
    __shared__ float shlds[CPADIN];

    const int tid = threadIdx.x;
    for (int i = tid; i < OUTC * CPADIN; i += 256) {
        const int o = i / CPADIN, c = i % CPADIN;
        wlds[i] = (_Float16)((c < CIN) ? W[o * CIN + c] : 0.0f);
    }
    for (int i = tid; i < OUTC; i += 256) blds[i] = bias[i];
    if (APPLY_BN) {
        for (int i = tid; i < CPADIN; i += 256) {
            sclds[i] = in_scale[i];
            shlds[i] = in_shift[i];
        }
    }
    __syncthreads();

    const int wave    = tid >> 5;
    const int lane    = tid & 31;
    const int halfsel = lane >> 4;                 // 0 or 1
    const int l16     = lane & 15;
    const long long pos = (long long)blockIdx.x * 128 + wave * 16 + l16;

    v8f acc[MT];
#pragma unroll
    for (int mt = 0; mt < MT; ++mt) { v8f z = {}; acc[mt] = z; }

    // hint-prefetch the next block's B operand rows
    __builtin_prefetch(Xin + (pos + 128) * CPADIN, 0, 1);

#pragma unroll
    for (int kc = 0; kc < KC; ++kc) {
        // ---- B fragment: 32x16 (KxN), lane = column n, 16 contiguous K ----
        const int cb = kc * 32 + halfsel * 16;
        v16h bfrag = *(const v16h*)(Xin + pos * CPADIN + cb);
        if (APPLY_BN) {
#pragma unroll
            for (int e = 0; e < 16; ++e) {
                const float v = (float)bfrag[e] * sclds[cb + e] + shlds[cb + e];
                bfrag[e] = (_Float16)fmaxf(v, 0.0f);      // BN + ReLU fused into load
            }
        }
        // ---- A fragments: weights, 16x32 (MxK), 16 contiguous K per lane half ----
        const int kb = halfsel * 16;
#pragma unroll
        for (int mt = 0; mt < MT; ++mt) {
            const _Float16* wp = &wlds[(mt * 16 + l16) * CPADIN + kc * 32 + kb];
            const v8h alo = *(const v8h*)(wp);            // K = kb .. kb+7
            const v8h ahi = *(const v8h*)(wp + 8);        // K = kb+8 .. kb+15
            const v16h afrag = __builtin_shufflevector(
                alo, ahi, 0, 1, 2, 3, 4, 5, 6, 7, 8, 9, 10, 11, 12, 13, 14, 15);
            acc[mt] = __builtin_amdgcn_wmma_f32_16x16x32_f16(
                false, afrag, false, bfrag, (short)0, acc[mt], false, false);
        }
    }

    // ---- epilogue: bias, f16 store, per-channel sum/sumsq via shuffles ----
#pragma unroll
    for (int mt = 0; mt < MT; ++mt) {
#pragma unroll
        for (int v = 0; v < 8; ++v) {
            const int o = mt * 16 + halfsel * 8 + v;      // C/D layout: row = v + half*8
            const float y = acc[mt][v] + blds[o];
            Yout[pos * OUTC + o] = (_Float16)y;
            float s = y, q = y * y;
#pragma unroll
            for (int off = 1; off < 16; off <<= 1) {
                s += __shfl_xor(s, off);
                q += __shfl_xor(q, off);
            }
            if (l16 == 0) {
                atomicAdd(&sum_out[o], s);
                atomicAdd(&sq_out[o], q);
            }
        }
    }
}

// =====================================================================
// 4) fold sums into BN scale/shift:  yhat*g+be = y*scale + shift
// =====================================================================
__global__ void bn_finalize_kernel(const float* __restrict__ sum,
                                   const float* __restrict__ sq,
                                   const float* __restrict__ g,
                                   const float* __restrict__ be,
                                   float* __restrict__ scale,
                                   float* __restrict__ shift,
                                   int C, float invP) {
    const int i = threadIdx.x;
    if (i < C) {
        const float mean = sum[i] * invP;
        const float var  = sq[i] * invP - mean * mean;
        const float rs   = rsqrtf(var + 1e-5f);
        const float sc   = rs * g[i];
        scale[i] = sc;
        shift[i] = be[i] - mean * sc;
    }
}

// =====================================================================
// 5) BN3 + ReLU + max over nsample, write [B,128,S] to d_out
// =====================================================================
__global__ void __launch_bounds__(128)
maxpool_kernel(const _Float16* __restrict__ Y3,
               const float* __restrict__ scale3,
               const float* __restrict__ shift3,
               float* __restrict__ out) {
    const int bs = blockIdx.x;                     // b*NPOINT + s
    const int o  = threadIdx.x;                    // 0..127
    const float sc = scale3[o], sh = shift3[o];
    const _Float16* p = Y3 + (size_t)bs * NSAMPLE * 128 + o;
    float m = -1e30f;
#pragma unroll 8
    for (int k = 0; k < NSAMPLE; ++k) {
        const float v = (float)p[k * 128] * sc + sh;
        m = fmaxf(m, v);
    }
    const int b = bs >> 10, s = bs & 1023;
    out[((size_t)b * 128 + o) * 1024 + s] = fmaxf(m, 0.0f);
}

// =====================================================================
extern "C" void kernel_launch(void* const* d_in, const int* in_sizes, int n_in,
                              void* d_out, int out_size, void* d_ws, size_t ws_size,
                              hipStream_t stream) {
    const float* xyz    = (const float*)d_in[0];
    const float* points = (const float*)d_in[1];
    const float* w1 = (const float*)d_in[2];
    const float* b1 = (const float*)d_in[3];
    const float* g1 = (const float*)d_in[4];
    const float* be1 = (const float*)d_in[5];
    const float* w2 = (const float*)d_in[6];
    const float* b2 = (const float*)d_in[7];
    const float* g2 = (const float*)d_in[8];
    const float* be2 = (const float*)d_in[9];
    const float* w3 = (const float*)d_in[10];
    const float* b3 = (const float*)d_in[11];
    const float* g3 = (const float*)d_in[12];
    const float* be3 = (const float*)d_in[13];

    float* out = (float*)d_out;
    float* new_xyz = out;                                  // [16,1024,3]
    float* new_points_out = out + (size_t)Bb * NPOINT * 3; // [16,128,1024]

    // ---- workspace carve-up (all offsets 256B / 32B aligned) ----
    char* ws = (char*)d_ws;
    int*   fps_idx = (int*)ws;                                   //  64 KiB
    float* stats   = (float*)(ws + 65536);                       //   4 KiB
    _Float16* X0 = (_Float16*)(ws + 69632);                      //  96 MiB  [P][96]
    _Float16* Y1 = (_Float16*)(ws + 69632 + Ppos * CPAD1 * 2);   //  64 MiB  [P][64]
    _Float16* Y2 = (_Float16*)((char*)Y1 + Ppos * 64 * 2);       //  64 MiB  [P][64]
    _Float16* Y3 = (_Float16*)((char*)Y2 + Ppos * 64 * 2);       // 128 MiB  [P][128]

    float* sum1 = stats +   0; float* sq1 = stats +  64;
    float* sum2 = stats + 128; float* sq2 = stats + 192;
    float* sum3 = stats + 256; float* sq3 = stats + 384;
    float* scale1 = stats + 512; float* shift1 = stats + 576;
    float* scale2 = stats + 640; float* shift2 = stats + 704;
    float* scale3 = stats + 768; float* shift3 = stats + 896;

    const float invP = 1.0f / (float)Ppos;
    const int convBlocks = (int)(Ppos / 128);                    // 4096

    zero_stats_kernel<<<2, 256, 0, stream>>>(stats);

    fps_kernel<<<Bb, 1024, 0, stream>>>(xyz, fps_idx, new_xyz);

    ball_group_kernel<<<(Bb * NPOINT) / 8, 256, 0, stream>>>(xyz, points, new_xyz, X0);

    conv_bn_wmma_kernel<64, CPAD1, 3, false><<<convBlocks, 256, 0, stream>>>(
        X0, w1, 67, b1, nullptr, nullptr, sum1, sq1, Y1);
    bn_finalize_kernel<<<1, 128, 0, stream>>>(sum1, sq1, g1, be1, scale1, shift1, 64, invP);

    conv_bn_wmma_kernel<64, 64, 2, true><<<convBlocks, 256, 0, stream>>>(
        Y1, w2, 64, b2, scale1, shift1, sum2, sq2, Y2);
    bn_finalize_kernel<<<1, 128, 0, stream>>>(sum2, sq2, g2, be2, scale2, shift2, 64, invP);

    conv_bn_wmma_kernel<128, 64, 2, true><<<convBlocks, 256, 0, stream>>>(
        Y2, w3, 64, b3, scale2, shift2, sum3, sq3, Y3);
    bn_finalize_kernel<<<1, 128, 0, stream>>>(sum3, sq3, g3, be3, scale3, shift3, 128, invP);

    maxpool_kernel<<<Bb * NPOINT, 128, 0, stream>>>(Y3, scale3, shift3, new_points_out);
}
